// QGNN_Rot2_PermInv_Layer_28217935135257
// MI455X (gfx1250) — matrix operand order repaired and checked by reading, not compiled, not run on hardware
//
#include <hip/hip_runtime.h>
#include <hip/hip_bf16.h>

typedef __attribute__((ext_vector_type(16))) _Float16 v16h;
typedef __attribute__((ext_vector_type(8)))  float    v8f;

#define PI_F     3.14159265358979323846f
#define TWO_PI_F 6.28318530717958647692f

union HF16 { v16h v; unsigned u[8]; };

__device__ __forceinline__ float silu_f(float x) {
  return x / (1.0f + __expf(-x));
}

// Assemble 16-bit A fragment (16x32, MxK): lane L holds row M=L&15;
// VGPR v in 0..3 -> K = (L>>4)*8 + 2v, 2v+1 ; v in 4..7 -> K = 16 + (L>>4)*8 + ...
__device__ __forceinline__ v16h load_a32(const _Float16* row, int kb8) {
  HF16 f;
#pragma unroll
  for (int v = 0; v < 4; ++v) {
    f.u[v]     = *(const unsigned*)(row + kb8 + 2 * v);
    f.u[v + 4] = *(const unsigned*)(row + 16 + kb8 + 2 * v);
  }
  return f.v;
}

// Assemble 16-bit B fragment (32x16, KxN): lane L holds column N=L&15;
// VGPR v in 0..3 -> K = (L>>4)*8 + 2v, 2v+1 ; v in 4..7 -> K = 16 + (L>>4)*8 + ...  `col` points at W[n][k_offset] (K contiguous).
__device__ __forceinline__ v16h load_b32(const _Float16* col, int kb16) {
  HF16 f;
#pragma unroll
  for (int v = 0; v < 4; ++v) {
    f.u[v]     = *(const unsigned*)(col + (kb16 >> 1) + 2 * v);
    f.u[v + 4] = *(const unsigned*)(col + 16 + (kb16 >> 1) + 2 * v);
  }
  return f.v;
}

#define WAVES 4
#define TPW   4
#define EDGES_PER_BLOCK (WAVES * TPW * 16)

__global__ __launch_bounds__(128) void qgnn_edge_kernel(
    const float* __restrict__ hre, const float* __restrict__ him,
    const int*   __restrict__ ei,
    const float* __restrict__ npar, const float* __restrict__ epar,
    const float* __restrict__ Wa,  const float* __restrict__ ba,
    const float* __restrict__ Wr1, const float* __restrict__ br1,
    const float* __restrict__ Wr2, const float* __restrict__ br2,
    const float* __restrict__ Wr3, const float* __restrict__ br3,
    const float* __restrict__ Wi1, const float* __restrict__ bi1,
    const float* __restrict__ Wi2, const float* __restrict__ bi2,
    const float* __restrict__ Wi3, const float* __restrict__ bi3,
    float* __restrict__ agg, int Etot)
{
  // Weights transposed (N-major, K contiguous) so B-fragment dwords are single aligned DS loads.
  // sW1[0]: real W1 in K rows 0..13 (rest 0); sW1[1]: imag W1 in K rows 16..29 (rest 0).
  __shared__ _Float16 sW1[2][64][32];
  __shared__ _Float16 sW2[2][64][64];
  __shared__ _Float16 sW3[2][16][64];
  __shared__ float sB1[2][64], sB2[2][64], sB3[2][4];
  __shared__ float sWa[8][3], sBa[3];
  // per-wave scratch
  __shared__ _Float16 sX [WAVES][16][32];   // cols 0..13 real feats, 16..29 imag feats
  __shared__ _Float16 sH1[WAVES][16][64];
  __shared__ _Float16 sH2[WAVES][16][64];
  __shared__ float    sAr[WAVES][16][4];
  __shared__ float    sAi[WAVES][16][4];

  const int tid = threadIdx.x;
  for (int idx = tid; idx < 64 * 32; idx += 128) {
    int n = idx >> 5, k = idx & 31;
    sW1[0][n][k] = (_Float16)((k < 14) ? Wr1[k * 64 + n] : 0.0f);
    sW1[1][n][k] = (_Float16)((k >= 16 && k < 30) ? Wi1[(k - 16) * 64 + n] : 0.0f);
  }
  for (int idx = tid; idx < 64 * 64; idx += 128) {
    int n = idx >> 6, k = idx & 63;
    sW2[0][n][k] = (_Float16)Wr2[k * 64 + n];
    sW2[1][n][k] = (_Float16)Wi2[k * 64 + n];
  }
  for (int idx = tid; idx < 16 * 64; idx += 128) {
    int n = idx >> 6, k = idx & 63;
    sW3[0][n][k] = (_Float16)((n < 4) ? Wr3[k * 4 + n] : 0.0f);
    sW3[1][n][k] = (_Float16)((n < 4) ? Wi3[k * 4 + n] : 0.0f);
  }
  if (tid < 64) { sB1[0][tid] = br1[tid]; sB1[1][tid] = bi1[tid];
                  sB2[0][tid] = br2[tid]; sB2[1][tid] = bi2[tid]; }
  if (tid < 4)  { sB3[0][tid] = br3[tid]; sB3[1][tid] = bi3[tid]; }
  if (tid < 24) sWa[tid / 3][tid % 3] = Wa[tid];
  if (tid < 3)  sBa[tid] = ba[tid];
  __syncthreads();

  const int w    = tid >> 5;
  const int lane = tid & 31;
  const int m16  = lane & 15;
  const int lh   = lane >> 4;       // lane half
  const int kb8  = lh * 8;
  const int kb16 = lh * 16;

  for (int t = 0; t < TPW; ++t) {
    const long base = (long)blockIdx.x * EDGES_PER_BLOCK + (long)w * (TPW * 16) + (long)t * 16;
    if (base >= Etot) break;        // uniform across the wave

    // ---- gather: lanes 0..15 fill the real feature block, 16..31 the imag block ----
    {
      int e     = (int)base + m16;
      bool vld  = (e < Etot);
      int ec    = vld ? e : (Etot - 1);
      int snd   = ei[ec];
      int rcv   = ei[Etot + ec];
      const float* hsrc = lh ? him : hre;
      _Float16* dst = &sX[w][m16][lh * 16];
      dst[0]  = (_Float16)hsrc[snd * 4 + 0];
      dst[1]  = (_Float16)hsrc[snd * 4 + 1];
      dst[2]  = (_Float16)hsrc[snd * 4 + 2];
      dst[3]  = (_Float16)hsrc[snd * 4 + 3];
      dst[4]  = (_Float16)hsrc[rcv * 4 + 0];
      dst[5]  = (_Float16)hsrc[rcv * 4 + 1];
      dst[6]  = (_Float16)hsrc[rcv * 4 + 2];
      dst[7]  = (_Float16)hsrc[rcv * 4 + 3];
      dst[8]  = (_Float16)npar[snd * 2 + 0];
      dst[9]  = (_Float16)npar[snd * 2 + 1];
      dst[10] = (_Float16)npar[rcv * 2 + 0];
      dst[11] = (_Float16)npar[rcv * 2 + 1];
      // reference's edge_params[edge_index].reshape(E,-1) quirk: flat pairs (2e, 2e+1)
      dst[12] = (_Float16)epar[ei[2 * ec + 0]];
      dst[13] = (_Float16)epar[ei[2 * ec + 1]];
      dst[14] = (_Float16)0.0f;
      dst[15] = (_Float16)0.0f;
    }
    __builtin_amdgcn_wave_barrier();   // DS ops are in-order per wave; stop compiler reordering

    // One A fragment serves both MLPs (real feats at K<16, imag feats at K>=16;
    // weight tiles are zero on the complementary K range).
    const v16h aIn = load_a32(&sX[w][m16][0], kb8);

#pragma unroll
    for (int p = 0; p < 2; ++p) {     // p=0: real MLP, p=1: imag MLP
      // ---- layer 1: 16x64 = A(16x32) x W1(32x64), 4 N-tiles ----
#pragma unroll
      for (int nt = 0; nt < 4; ++nt) {
        v16h b = load_b32(&sW1[p][nt * 16 + m16][0], kb16);
        v8f acc = {};
        acc = __builtin_amdgcn_wmma_f32_16x16x32_f16(false, aIn, false, b,
                                                     (short)0, acc, false, false);
        const int n = nt * 16 + m16;
        const float bias = sB1[p][n];
#pragma unroll
        for (int r = 0; r < 8; ++r) {
          float x = silu_f(acc[r] + bias);
          sH1[w][lh * 8 + r][n] = (_Float16)x;
        }
      }
      __builtin_amdgcn_wave_barrier();

      // ---- layer 2: 16x64 = H1(16x64) x W2(64x64), 2 K-steps x 4 N-tiles ----
      v16h a2[2];
#pragma unroll
      for (int ks = 0; ks < 2; ++ks)
        a2[ks] = load_a32(&sH1[w][m16][32 * ks], kb8);
#pragma unroll
      for (int nt = 0; nt < 4; ++nt) {
        v8f acc = {};
#pragma unroll
        for (int ks = 0; ks < 2; ++ks) {
          v16h b = load_b32(&sW2[p][nt * 16 + m16][32 * ks], kb16);
          acc = __builtin_amdgcn_wmma_f32_16x16x32_f16(false, a2[ks], false, b,
                                                       (short)0, acc, false, false);
        }
        const int n = nt * 16 + m16;
        const float bias = sB2[p][n];
#pragma unroll
        for (int r = 0; r < 8; ++r) {
          float x = silu_f(acc[r] + bias);
          sH2[w][lh * 8 + r][n] = (_Float16)x;
        }
      }
      __builtin_amdgcn_wave_barrier();

      // ---- layer 3: 16x4 (padded to 16) = H2(16x64) x W3(64x16), 2 K-steps ----
      v8f acc3 = {};
#pragma unroll
      for (int ks = 0; ks < 2; ++ks) {
        v16h a3 = load_a32(&sH2[w][m16][32 * ks], kb8);
        v16h b  = load_b32(&sW3[p][m16][32 * ks], kb16);
        acc3 = __builtin_amdgcn_wmma_f32_16x16x32_f16(false, a3, false, b,
                                                      (short)0, acc3, false, false);
      }
      if (m16 < 4) {
        const float bias = sB3[p][m16];
        float (*dst)[4] = p ? sAi[w] : sAr[w];
#pragma unroll
        for (int r = 0; r < 8; ++r)
          dst[lh * 8 + r][m16] = acc3[r] + bias;
      }
      __builtin_amdgcn_wave_barrier();
    }

    // ---- angle head (8x3) + tanh + segment-sum scatter ----
    if (lh == 0) {
      int e = (int)base + m16;
      if (e < Etot) {
        int rcv = ei[Etot + e];
        float cat[8];
#pragma unroll
        for (int j = 0; j < 4; ++j) {           // concat([A_imag, A_real])
          cat[j]     = sAi[w][m16][j];
          cat[4 + j] = sAr[w][m16][j];
        }
#pragma unroll
        for (int k = 0; k < 3; ++k) {
          float v = sBa[k];
#pragma unroll
          for (int j = 0; j < 8; ++j) v += cat[j] * sWa[j][k];
          atomicAdd(&agg[rcv * 3 + k], tanhf(v) * PI_F);
        }
      }
    }
    __builtin_amdgcn_wave_barrier();
  }
}

__device__ __forceinline__ float wrap_pi(float x) {
  float r = fmodf(x + PI_F, TWO_PI_F);
  if (r < 0.0f) r += TWO_PI_F;
  return r - PI_F;
}
__device__ __forceinline__ float2 cmul(float2 a, float2 b) {
  return make_float2(a.x * b.x - a.y * b.y, a.x * b.y + a.y * b.x);
}
__device__ __forceinline__ float2 cmulc(float2 a, float2 b) {  // a * conj(b)
  return make_float2(a.x * b.x + a.y * b.y, a.y * b.x - a.x * b.y);
}
__device__ __forceinline__ float2 cadd(float2 a, float2 b) {
  return make_float2(a.x + b.x, a.y + b.y);
}

__global__ __launch_bounds__(256) void qgnn_node_kernel(
    const float* __restrict__ hre, const float* __restrict__ him,
    const float* __restrict__ agg, float* __restrict__ out, int Nn)
{
  int i = blockIdx.x * blockDim.x + threadIdx.x;
  if (i >= Nn) return;
  float a = wrap_pi(agg[i * 3 + 0]);
  float b = wrap_pi(agg[i * 3 + 1]);
  float c = wrap_pi(agg[i * 3 + 2]);
  float al = 0.5f * (a + c), dl = 0.5f * (a - c);
  float cb = cosf(0.5f * b), sb = sinf(0.5f * b);
  float ca = cosf(al), sa = sinf(al);
  float cd = cosf(dl), sd = sinf(dl);
  float2 U00 = make_float2( ca * cb, -sa * cb);   //  e^{-i al} cb
  float2 U01 = make_float2(-cd * sb,  sd * sb);   // -e^{-i dl} sb
  float2 U10 = make_float2( cd * sb,  sd * sb);   //  e^{+i dl} sb
  float2 U11 = make_float2( ca * cb,  sa * cb);   //  e^{+i al} cb
  float2 h00 = make_float2(hre[i * 4 + 0], him[i * 4 + 0]);
  float2 h01 = make_float2(hre[i * 4 + 1], him[i * 4 + 1]);
  float2 h10 = make_float2(hre[i * 4 + 2], him[i * 4 + 2]);
  float2 h11 = make_float2(hre[i * 4 + 3], him[i * 4 + 3]);
  // T = h * U^dagger ; T[r][c] = sum_k h[r][k] * conj(U[c][k])
  float2 T00 = cadd(cmulc(h00, U00), cmulc(h01, U01));
  float2 T01 = cadd(cmulc(h00, U10), cmulc(h01, U11));
  float2 T10 = cadd(cmulc(h10, U00), cmulc(h11, U01));
  float2 T11 = cadd(cmulc(h10, U10), cmulc(h11, U11));
  // O = U * T
  float2 O00 = cadd(cmul(U00, T00), cmul(U01, T10));
  float2 O01 = cadd(cmul(U00, T01), cmul(U01, T11));
  float2 O10 = cadd(cmul(U10, T00), cmul(U11, T10));
  float2 O11 = cadd(cmul(U10, T01), cmul(U11, T11));
  float* o = out + (size_t)i * 8;    // complex64 interleaved, row-major 2x2
  o[0] = O00.x; o[1] = O00.y;
  o[2] = O01.x; o[3] = O01.y;
  o[4] = O10.x; o[5] = O10.y;
  o[6] = O11.x; o[7] = O11.y;
}

__global__ void zero_f32_kernel(float* __restrict__ p, int n) {
  int i = blockIdx.x * blockDim.x + threadIdx.x;
  if (i < n) p[i] = 0.0f;
}

extern "C" void kernel_launch(void* const* d_in, const int* in_sizes, int n_in,
                              void* d_out, int out_size, void* d_ws, size_t ws_size,
                              hipStream_t stream) {
  (void)n_in; (void)out_size; (void)ws_size;
  const float* hre  = (const float*)d_in[0];
  const float* him  = (const float*)d_in[1];
  const int*   ei   = (const int*)  d_in[2];
  const float* npar = (const float*)d_in[3];
  const float* epar = (const float*)d_in[4];
  const float* Wa   = (const float*)d_in[5];
  const float* ba   = (const float*)d_in[6];
  const float* Wr1  = (const float*)d_in[7];  const float* br1 = (const float*)d_in[8];
  const float* Wr2  = (const float*)d_in[9];  const float* br2 = (const float*)d_in[10];
  const float* Wr3  = (const float*)d_in[11]; const float* br3 = (const float*)d_in[12];
  const float* Wi1  = (const float*)d_in[13]; const float* bi1 = (const float*)d_in[14];
  const float* Wi2  = (const float*)d_in[15]; const float* bi2 = (const float*)d_in[16];
  const float* Wi3  = (const float*)d_in[17]; const float* bi3 = (const float*)d_in[18];

  const int Nn   = in_sizes[0] / 4;   // h_i_real is (N,2,2)
  const int Etot = in_sizes[2] / 2;   // edge_index is (2,E)
  float* agg = (float*)d_ws;          // N x 3 accumulator

  const int zn = Nn * 3;
  zero_f32_kernel<<<(zn + 255) / 256, 256, 0, stream>>>(agg, zn);

  const int eblocks = (Etot + EDGES_PER_BLOCK - 1) / EDGES_PER_BLOCK;
  qgnn_edge_kernel<<<eblocks, 128, 0, stream>>>(
      hre, him, ei, npar, epar, Wa, ba,
      Wr1, br1, Wr2, br2, Wr3, br3,
      Wi1, bi1, Wi2, bi2, Wi3, bi3, agg, Etot);

  qgnn_node_kernel<<<(Nn + 255) / 256, 256, 0, stream>>>(hre, him, agg, (float*)d_out, Nn);
}
